// EncoderLayer_31147102830898
// MI455X (gfx1250) — matrix operand order repaired and checked by reading, not compiled, not run on hardware
//
#include <hip/hip_runtime.h>
#include <hip/hip_bf16.h>

typedef __attribute__((ext_vector_type(16))) __bf16 v16bf;
typedef __attribute__((ext_vector_type(8)))  float  v8f;

#define NTOK   16384   // B*S = 4*4096
#define DMODEL 1024
#define DFF    4096

__device__ __forceinline__ unsigned short f2bf(float f) {
  unsigned int u = __float_as_uint(f);
  u += 0x7FFFu + ((u >> 16) & 1u);         // round-to-nearest-even
  return (unsigned short)(u >> 16);
}
__device__ __forceinline__ float bf2f(unsigned short h) {
  return __uint_as_float(((unsigned int)h) << 16);
}

union Frag { v16bf v; uint4 q[2]; };

// ---------------------------------------------------------------------------
// fp32 -> bf16 transposing weight conversion (tiled through LDS, coalesced
// on both sides). in: R x C fp32 (row-major). out: C x R bf16, i.e.
// out[c*R + r] = bf16(in[r*C + c]).  Block (32,8), grid (C/32, R/32).
// ---------------------------------------------------------------------------
__global__ __launch_bounds__(256) void cvt_t_kernel(
    const float* __restrict__ in, unsigned short* __restrict__ out,
    int R, int C) {
  __shared__ float tile[32][33];
  const int tx = threadIdx.x;   // 0..31
  const int ty = threadIdx.y;   // 0..7
  const int r0 = blockIdx.y * 32;
  const int c0 = blockIdx.x * 32;
#pragma unroll
  for (int i = 0; i < 4; ++i)
    tile[ty + i * 8][tx] = in[(size_t)(r0 + ty + i * 8) * C + c0 + tx];
  __syncthreads();
#pragma unroll
  for (int i = 0; i < 4; ++i)
    out[(size_t)(c0 + ty + i * 8) * R + r0 + tx] = f2bf(tile[tx][ty + i * 8]);
}

// ---------------------------------------------------------------------------
// LayerNorm over rows of 1024, output bf16. One block (256 thr) per row.
// ---------------------------------------------------------------------------
__global__ __launch_bounds__(256) void ln_kernel(
    const float* __restrict__ x, const float* __restrict__ g,
    const float* __restrict__ b, unsigned short* __restrict__ out) {
  const int row = blockIdx.x;
  const int tid = threadIdx.x;
  const size_t base = (size_t)row * DMODEL;
  const float4 v = *(const float4*)(x + base + tid * 4);

  float s  = v.x + v.y + v.z + v.w;
  float sq = v.x * v.x + v.y * v.y + v.z * v.z + v.w * v.w;
  for (int o = 16; o; o >>= 1) {
    s  += __shfl_xor(s,  o);
    sq += __shfl_xor(sq, o);
  }
  __shared__ float rs[8], rq[8];
  const int lane = tid & 31, wv = tid >> 5;
  if (lane == 0) { rs[wv] = s; rq[wv] = sq; }
  __syncthreads();
  s = 0.f; sq = 0.f;
  for (int w = 0; w < 8; ++w) { s += rs[w]; sq += rq[w]; }

  const float mu   = s * (1.f / DMODEL);
  const float var  = sq * (1.f / DMODEL) - mu * mu;
  const float rstd = rsqrtf(var + 1e-5f);

  const int c = tid * 4;
  out[base + c + 0] = f2bf((v.x - mu) * rstd * g[c + 0] + b[c + 0]);
  out[base + c + 1] = f2bf((v.y - mu) * rstd * g[c + 1] + b[c + 1]);
  out[base + c + 2] = f2bf((v.z - mu) * rstd * g[c + 2] + b[c + 2]);
  out[base + c + 3] = f2bf((v.w - mu) * rstd * g[c + 3] + b[c + 3]);
}

// ---------------------------------------------------------------------------
// bf16 GEMM: C(MxN, fp32 acc) = A(MxK, row-major) * Bt(NxK, pre-transposed),
// via v_wmma_f32_16x16x32_bf16. Block: 256 thr = 8 waves; block tile 128x128;
// wave tile 32x64 (2x4 WMMA tiles). K step 32, double-buffered LDS; both A
// and Bt staged with b128 loads/stores (no scalar LDS scatter).
// EPI: 0 = store bf16; 1 = +bias, exact GELU, store bf16;
//      2 = +bias +residual, store fp32
// ---------------------------------------------------------------------------
template <int EPI>
__global__ __launch_bounds__(256) void gemm_bf16_kernel(
    const unsigned short* __restrict__ A, const unsigned short* __restrict__ Bt,
    const float* __restrict__ bias, const float* __restrict__ res,
    unsigned short* __restrict__ outb, float* __restrict__ outf,
    int M, int N, int K) {
  __shared__ __align__(16) unsigned short As[2][128][40];
  __shared__ __align__(16) unsigned short Bs[2][128][40];   // [n][k]

  const int tid = threadIdx.x;
  const int bm  = blockIdx.y * 128;
  const int bn  = blockIdx.x * 128;
  const int lane = tid & 31;
  const int wid  = tid >> 5;
  const int wm = wid & 3;           // 4 waves along M
  const int wn = wid >> 2;          // 2 waves along N
  const int lm = lane & 15;
  const int hi = lane >> 4;         // half-wave id

  // staging: both tiles are 128 rows x 32 k halves; 32B (16 halves) per thread
  const int srow  = tid >> 1;
  const int shalf = tid & 1;

  const unsigned short* gA = A  + (size_t)(bm + srow) * K + shalf * 16;
  const unsigned short* gB = Bt + (size_t)(bn + srow) * K + shalf * 16;

  v8f acc[2][4];
  const v8f vzero = {0.f, 0.f, 0.f, 0.f, 0.f, 0.f, 0.f, 0.f};
#pragma unroll
  for (int i = 0; i < 2; ++i)
#pragma unroll
    for (int j = 0; j < 4; ++j) acc[i][j] = vzero;

  const int nk = K >> 5;

  uint4 ra0 = ((const uint4*)gA)[0];
  uint4 ra1 = ((const uint4*)gA)[1];
  uint4 rb0 = ((const uint4*)gB)[0];
  uint4 rb1 = ((const uint4*)gB)[1];

  {  // stage chunk 0 into buffer 0
    *(uint4*)&As[0][srow][shalf * 16]     = ra0;
    *(uint4*)&As[0][srow][shalf * 16 + 8] = ra1;
    *(uint4*)&Bs[0][srow][shalf * 16]     = rb0;
    *(uint4*)&Bs[0][srow][shalf * 16 + 8] = rb1;
  }
  __syncthreads();

  for (int kc = 0; kc < nk; ++kc) {
    const int buf = kc & 1;
    const bool more = (kc + 1) < nk;
    if (more) {  // prefetch next chunk into registers
      const unsigned short* pA = gA + (size_t)(kc + 1) * 32;
      const unsigned short* pB = gB + (size_t)(kc + 1) * 32;
      ra0 = ((const uint4*)pA)[0];
      ra1 = ((const uint4*)pA)[1];
      rb0 = ((const uint4*)pB)[0];
      rb1 = ((const uint4*)pB)[1];
    }

    // WMMA fragment loads per ISA layout:
    //   A lane<16: k = {0..7, 16..23}; lane>=16: k = {8..15, 24..31}
    //   B lane<16: k = {0..7, 16..23}; lane>=16: k = {8..15, 24..31}
    Frag af[2];
#pragma unroll
    for (int tm = 0; tm < 2; ++tm) {
      const int r = wm * 32 + tm * 16 + lm;
      af[tm].q[0] = *(const uint4*)&As[buf][r][hi * 8];
      af[tm].q[1] = *(const uint4*)&As[buf][r][hi * 8 + 16];
    }
#pragma unroll
    for (int tn = 0; tn < 4; ++tn) {
      const int c = wn * 64 + tn * 16 + lm;
      Frag bfr;
      bfr.q[0] = *(const uint4*)&Bs[buf][c][hi * 8];
      bfr.q[1] = *(const uint4*)&Bs[buf][c][hi * 8 + 16];
#pragma unroll
      for (int tm = 0; tm < 2; ++tm) {
        acc[tm][tn] = __builtin_amdgcn_wmma_f32_16x16x32_bf16(
            false, af[tm].v, false, bfr.v, (short)0, acc[tm][tn], false, false);
      }
    }

    if (more) {  // stage prefetched chunk into the other buffer
      const int nb = buf ^ 1;
      *(uint4*)&As[nb][srow][shalf * 16]     = ra0;
      *(uint4*)&As[nb][srow][shalf * 16 + 8] = ra1;
      *(uint4*)&Bs[nb][srow][shalf * 16]     = rb0;
      *(uint4*)&Bs[nb][srow][shalf * 16 + 8] = rb1;
    }
    __syncthreads();
  }

  // Epilogue. C/D layout: vgpr r -> m = r + 8*hi; n = lane&15.
  const int mro = bm + wm * 32 + hi * 8;
  const int nco = bn + wn * 64 + lm;
#pragma unroll
  for (int tm = 0; tm < 2; ++tm) {
#pragma unroll
    for (int tn = 0; tn < 4; ++tn) {
      const int nn = nco + tn * 16;
      float bv = 0.f;
      if constexpr (EPI == 1 || EPI == 2) bv = bias[nn];
#pragma unroll
      for (int r = 0; r < 8; ++r) {
        const int mm = mro + tm * 16 + r;
        float v = acc[tm][tn][r];
        if constexpr (EPI == 0) {
          outb[(size_t)mm * N + nn] = f2bf(v);
        } else if constexpr (EPI == 1) {
          v += bv;
          v = 0.5f * v * (1.f + erff(v * 0.70710678118654752f));
          outb[(size_t)mm * N + nn] = f2bf(v);
        } else {
          outf[(size_t)mm * N + nn] = res[(size_t)mm * N + nn] + v + bv;
        }
      }
    }
  }
}

// ---------------------------------------------------------------------------
// Per-token attention on the packed qkv buffer (row stride 3072):
// q,k,v rows (hd=64 x H=16); scores 64x64 contracting over H; softmax over j;
// out = probs @ v. One 64-thread block per token, fp32 math.
// q[b,s,d,h] = row[h*64 + d] (split_heads + swapaxes in the reference).
// ---------------------------------------------------------------------------
__global__ __launch_bounds__(64) void attn_kernel(
    const unsigned short* __restrict__ QKV, const float* __restrict__ unc,
    unsigned short* __restrict__ O) {
  __shared__ float qs[64][17];
  __shared__ float ks[64][17];
  __shared__ float vs[64][17];
  const int i = threadIdx.x;  // head-dim index 0..63
  const size_t bin  = (size_t)blockIdx.x * 3072;
  const size_t bout = (size_t)blockIdx.x * DMODEL;
#pragma unroll
  for (int h = 0; h < 16; ++h) {
    qs[i][h] = bf2f(QKV[bin +        h * 64 + i]);
    ks[i][h] = bf2f(QKV[bin + 1024 + h * 64 + i]);
    vs[i][h] = bf2f(QKV[bin + 2048 + h * 64 + i]);
  }
  float su = 0.f;
#pragma unroll
  for (int h = 0; h < 16; ++h) su += unc[h];
  const float inv = 1.f / (8.f * su);  // sqrt(64) * sum(u)
  __syncthreads();

  float s[64];
  float mx = -3.4e38f;
  for (int j = 0; j < 64; ++j) {
    float d = 0.f;
#pragma unroll
    for (int h = 0; h < 16; ++h) d += qs[i][h] * ks[j][h];
    d *= inv;
    s[j] = d;
    mx = fmaxf(mx, d);
  }
  float sum = 0.f;
  for (int j = 0; j < 64; ++j) { s[j] = expf(s[j] - mx); sum += s[j]; }
  const float rcp = 1.f / sum;
#pragma unroll
  for (int h = 0; h < 16; ++h) {
    float a = 0.f;
    for (int j = 0; j < 64; ++j) a += s[j] * vs[j][h];
    O[bout + h * 64 + i] = f2bf(a * rcp);
  }
}

// ---------------------------------------------------------------------------
extern "C" void kernel_launch(void* const* d_in, const int* in_sizes, int n_in,
                              void* d_out, int out_size, void* d_ws, size_t ws_size,
                              hipStream_t stream) {
  const float* x    = (const float*)d_in[0];
  const float* unc  = (const float*)d_in[1];
  const float* Wq   = (const float*)d_in[2];
  const float* Wk   = (const float*)d_in[3];
  const float* Wv   = (const float*)d_in[4];
  const float* Wo   = (const float*)d_in[5];
  const float* bo   = (const float*)d_in[6];
  const float* ln1g = (const float*)d_in[7];
  const float* ln1b = (const float*)d_in[8];
  const float* ln2g = (const float*)d_in[9];
  const float* ln2b = (const float*)d_in[10];
  const float* W1   = (const float*)d_in[11];
  const float* b1   = (const float*)d_in[12];
  const float* W2   = (const float*)d_in[13];
  const float* b2   = (const float*)d_in[14];
  float* out = (float*)d_out;
  char* ws = (char*)d_ws;

  size_t o = 0;
  auto take = [&](size_t bytes) {
    size_t r = o;
    o += (bytes + 255) & ~(size_t)255;
    return r;
  };
  unsigned short* t1   = (unsigned short*)(ws + take((size_t)NTOK * DMODEL * 2));
  unsigned short* qkv  = (unsigned short*)(ws + take((size_t)NTOK * 3072 * 2));
  unsigned short* ab   = (unsigned short*)(ws + take((size_t)NTOK * DMODEL * 2));
  float*          x1   = (float*)(ws + take((size_t)NTOK * DMODEL * 4));
  unsigned short* t2   = (unsigned short*)(ws + take((size_t)NTOK * DMODEL * 2));
  unsigned short* h1   = (unsigned short*)(ws + take((size_t)NTOK * DFF * 2));
  unsigned short* qkvT = (unsigned short*)(ws + take((size_t)3072 * DMODEL * 2));
  unsigned short* woT  = (unsigned short*)(ws + take((size_t)DMODEL * DMODEL * 2));
  unsigned short* w1T  = (unsigned short*)(ws + take((size_t)DMODEL * DFF * 2));
  unsigned short* w2T  = (unsigned short*)(ws + take((size_t)DMODEL * DFF * 2));

  // weights -> bf16, transposed to NxK; Wq/Wk/Wv packed into one 3072x1024
  const dim3 tb(32, 8);
  cvt_t_kernel<<<dim3(32, 32),  tb, 0, stream>>>(Wq, qkvT,               DMODEL, DMODEL);
  cvt_t_kernel<<<dim3(32, 32),  tb, 0, stream>>>(Wk, qkvT + 1024 * 1024, DMODEL, DMODEL);
  cvt_t_kernel<<<dim3(32, 32),  tb, 0, stream>>>(Wv, qkvT + 2048 * 1024, DMODEL, DMODEL);
  cvt_t_kernel<<<dim3(32, 32),  tb, 0, stream>>>(Wo, woT,                DMODEL, DMODEL);
  cvt_t_kernel<<<dim3(128, 32), tb, 0, stream>>>(W1, w1T, DMODEL, DFF);
  cvt_t_kernel<<<dim3(32, 128), tb, 0, stream>>>(W2, w2T, DFF, DMODEL);

  // LN1
  ln_kernel<<<NTOK, 256, 0, stream>>>(x, ln1g, ln1b, t1);

  // fused Q/K/V projection: (16384x1024) x (1024x3072) -> packed qkv (bf16)
  gemm_bf16_kernel<0><<<dim3(3072 / 128, NTOK / 128), 256, 0, stream>>>(
      t1, qkvT, nullptr, nullptr, qkv, nullptr, NTOK, 3072, DMODEL);

  // per-token attention
  attn_kernel<<<NTOK, 64, 0, stream>>>(qkv, unc, ab);

  // x1 = x + attn @ Wo + bo
  gemm_bf16_kernel<2><<<dim3(DMODEL / 128, NTOK / 128), 256, 0, stream>>>(
      ab, woT, bo, x, nullptr, x1, NTOK, DMODEL, DMODEL);

  // LN2
  ln_kernel<<<NTOK, 256, 0, stream>>>(x1, ln2g, ln2b, t2);

  // h1 = gelu(t2 @ W1 + b1)
  gemm_bf16_kernel<1><<<dim3(DFF / 128, NTOK / 128), 256, 0, stream>>>(
      t2, w1T, b1, nullptr, h1, nullptr, NTOK, DFF, DMODEL);

  // out = x1 + h1 @ W2 + b2
  gemm_bf16_kernel<2><<<dim3(DMODEL / 128, NTOK / 128), 256, 0, stream>>>(
      h1, w2T, b2, x1, nullptr, out, NTOK, DMODEL, DFF);
}
